// DecoderLayer_53893249630404
// MI455X (gfx1250) — matrix operand order repaired and checked by reading, not compiled, not run on hardware
//
#include <hip/hip_runtime.h>
#include <stdint.h>

// ---------------------------------------------------------------------------
// Problem constants (from reference)
// ---------------------------------------------------------------------------
#define BB   4
#define SS   1024
#define HH   16
#define HKV  4
#define DD   128
#define HID  2048
#define INTR 8192
#define NTOK (BB * SS)             // 4096 rows
#define QKVN ((HH + 2 * HKV) * DD) // 3072

typedef __attribute__((ext_vector_type(16))) __bf16 v16bf;
typedef __attribute__((ext_vector_type(8)))  float  v8f;

union Frag16 {
    v16bf v;
    uint32_t u[8];
    unsigned short s[16];
};

static __device__ __forceinline__ unsigned short f2bf(float f) {
    union { float f; uint32_t u; } x; x.f = f;
    uint32_t r = (x.u + 0x7FFFu + ((x.u >> 16) & 1u)) >> 16;
    return (unsigned short)r;
}
static __device__ __forceinline__ uint32_t f2bf_pack(float lo, float hi) {
    return (uint32_t)f2bf(lo) | ((uint32_t)f2bf(hi) << 16);
}
static __device__ __forceinline__ float bf2f(unsigned short s) {
    union { uint32_t u; float f; } x; x.u = ((uint32_t)s) << 16;
    return x.f;
}
static __device__ __forceinline__ v8f wmma_bf16(const Frag16& a, const Frag16& b, v8f c) {
    return __builtin_amdgcn_wmma_f32_16x16x32_bf16(false, a.v, false, b.v, (short)0, c, false, false);
}
// CDNA5 async global->LDS copy (16 bytes per lane), tracked by ASYNCcnt.
static __device__ __forceinline__ void async_b128(void* lds, const void* gmem) {
    uint32_t l = (uint32_t)(size_t)lds;
    uint64_t g = (uint64_t)(size_t)gmem;
    asm volatile("global_load_async_to_lds_b128 %0, %1, off" :: "v"(l), "v"(g) : "memory");
}
static __device__ __forceinline__ void wait_async0() {
    asm volatile("s_wait_asynccnt 0" ::: "memory");
}

// ---------------------------------------------------------------------------
// fp32 -> bf16 conversion (weights), vectorized 4-wide
// ---------------------------------------------------------------------------
__global__ void convert_bf16_kernel(const float4* __restrict__ in,
                                    uint32_t* __restrict__ out, size_t n4) {
    size_t stride = (size_t)gridDim.x * blockDim.x;
    for (size_t i = (size_t)blockIdx.x * blockDim.x + threadIdx.x; i < n4; i += stride) {
        float4 f = in[i];
        uint2 p;
        p.x = f2bf_pack(f.x, f.y);
        p.y = f2bf_pack(f.z, f.w);
        *(uint2*)(out + i * 2) = p;
    }
}

// ---------------------------------------------------------------------------
// RMSNorm: per row of HID=2048, output bf16
// ---------------------------------------------------------------------------
__global__ __launch_bounds__(256) void rmsnorm_kernel(const float* __restrict__ x,
                                                      const float* __restrict__ w,
                                                      unsigned short* __restrict__ out) {
    __shared__ float red[256];
    int row = blockIdx.x;
    int tid = threadIdx.x;
    const float* xr = x + (size_t)row * HID;
    float ss = 0.f;
    for (int i = tid; i < HID; i += 256) { float v = xr[i]; ss += v * v; }
    red[tid] = ss;
    __syncthreads();
    for (int s = 128; s > 0; s >>= 1) {
        if (tid < s) red[tid] += red[tid + s];
        __syncthreads();
    }
    float rs = rsqrtf(red[0] * (1.0f / HID) + 1e-6f);
    unsigned short* orow = out + (size_t)row * HID;
    for (int i = tid; i < HID; i += 256)
        orow[i] = f2bf(xr[i] * rs * w[i]);
}

// ---------------------------------------------------------------------------
// RoPE + QKV split: qkv fp32 [NTOK][3072] -> q/k (roped, bf16), v (bf16)
// ---------------------------------------------------------------------------
__global__ __launch_bounds__(256) void rope_split_kernel(const float* __restrict__ qkv,
                                                         const int* __restrict__ pos,
                                                         unsigned short* __restrict__ q,
                                                         unsigned short* __restrict__ k,
                                                         unsigned short* __restrict__ v) {
    const float LOG_THETA = 9.210340371976184f; // ln(10000)
    int t = blockIdx.x;            // token index 0..4095
    int s = t & (SS - 1);
    int p = pos[s];
    const float* row = qkv + (size_t)t * QKVN;
    int tid = threadIdx.x;
    for (int i = tid; i < HH * 64; i += 256) {
        int h = i >> 6, d = i & 63;
        float inv = __expf(-LOG_THETA * (float)d * (1.0f / 64.0f));
        float ang = (float)p * inv;
        float c = __cosf(ang), sn = __sinf(ang);
        float t1 = row[h * DD + d], t2 = row[h * DD + d + 64];
        size_t o = ((size_t)t * HH + h) * DD + d;
        q[o]      = f2bf(t1 * c - t2 * sn);
        q[o + 64] = f2bf(t2 * c + t1 * sn);
    }
    for (int i = tid; i < HKV * 64; i += 256) {
        int h = i >> 6, d = i & 63;
        float inv = __expf(-LOG_THETA * (float)d * (1.0f / 64.0f));
        float ang = (float)p * inv;
        float c = __cosf(ang), sn = __sinf(ang);
        float t1 = row[HH * DD + h * DD + d], t2 = row[HH * DD + h * DD + d + 64];
        size_t o = ((size_t)t * HKV + h) * DD + d;
        k[o]      = f2bf(t1 * c - t2 * sn);
        k[o + 64] = f2bf(t2 * c + t1 * sn);
    }
    for (int i = tid; i < HKV * DD; i += 256)
        v[(size_t)t * HKV * DD + i] = f2bf(row[(HH + HKV) * DD + i]);
}

// ---------------------------------------------------------------------------
// SwiGLU: gu bf16 [NTOK][2*INTR] -> act bf16 [NTOK][INTR]
// ---------------------------------------------------------------------------
__global__ void swiglu_kernel(const unsigned short* __restrict__ gu,
                              unsigned short* __restrict__ act) {
    size_t n = (size_t)NTOK * INTR;
    size_t stride = (size_t)gridDim.x * blockDim.x;
    for (size_t i = (size_t)blockIdx.x * blockDim.x + threadIdx.x; i < n; i += stride) {
        size_t r = i >> 13, c = i & (INTR - 1);
        float g = bf2f(gu[(r << 14) + c]);
        float u = bf2f(gu[(r << 14) + INTR + c]);
        float sg = g / (1.0f + __expf(-g));
        act[i] = f2bf(sg * u);
    }
}

// ---------------------------------------------------------------------------
// bf16 WMMA GEMM: C[M,N] = A[M,K] * B[K,N] (+resid), A/B bf16 row-major.
// Block tile 64x256, 256 threads = 8 waves (2x4), wave tile 32x64.
// K staged 64 deep via global_load_async_to_lds_b128 (ASYNCcnt).
// MODE: 0 = store f32, 1 = f32 + resid, 2 = store bf16.
// ---------------------------------------------------------------------------
template<int MODE>
__global__ __launch_bounds__(256) void gemm_bf16_kernel(const unsigned short* __restrict__ A,
                                                        const unsigned short* __restrict__ B,
                                                        int M, int N, int K,
                                                        const float* __restrict__ resid,
                                                        float* __restrict__ Cf,
                                                        unsigned short* __restrict__ Cb) {
    __shared__ unsigned short As[64][72];    // 64 x 64 (+8 pad)
    __shared__ unsigned short Bs[64][264];   // 64 x 256 (+8 pad)

    int tid  = threadIdx.x;
    int lane = tid & 31, wv = tid >> 5;
    int lm = lane & 15, lh = lane >> 4;
    int bm = blockIdx.y * 64, bn = blockIdx.x * 256;
    int wm = (wv >> 2) * 32, wn = (wv & 3) * 64;

    v8f acc[2][4];
    #pragma unroll
    for (int i = 0; i < 2; ++i)
        #pragma unroll
        for (int j = 0; j < 4; ++j) { v8f z = {0.f,0.f,0.f,0.f,0.f,0.f,0.f,0.f}; acc[i][j] = z; }

    int ar = tid >> 2, ac = (tid & 3) * 16;  // A: 64 rows x 4 groups of 16 bf16
    int br = tid >> 2, bc = (tid & 3) * 64;  // B: 64 rows x 4 groups of 64 bf16
    const unsigned short* aptr = A + (size_t)(bm + ar) * K + ac;
    const unsigned short* bptr = B + (size_t)br * N + bn + bc;

    for (int k0 = 0; k0 < K; k0 += 64) {
        // async stage A (2 x b128) and B (8 x b128), global -> LDS direct
        async_b128(&As[ar][ac],     aptr + k0);
        async_b128(&As[ar][ac + 8], aptr + k0 + 8);
        const unsigned short* bsrc = bptr + (size_t)k0 * N;
        #pragma unroll
        for (int i = 0; i < 8; ++i)
            async_b128(&Bs[br][bc + i * 8], bsrc + i * 8);
        if (k0 + 64 < K) {
            __builtin_prefetch(aptr + k0 + 64, 0, 1);
            __builtin_prefetch(bptr + (size_t)(k0 + 64) * N, 0, 1);
        }
        wait_async0();
        __syncthreads();

        #pragma unroll
        for (int ks = 0; ks < 64; ks += 32) {
            Frag16 fa[2], fb[4];
            #pragma unroll
            for (int tm = 0; tm < 2; ++tm) {
                int row = wm + tm * 16 + lm;
                #pragma unroll
                for (int i = 0; i < 8; ++i) {
                    int kk = ks + ((i < 4) ? (lh * 8 + 2 * i) : (16 + lh * 8 + 2 * (i - 4)));
                    fa[tm].u[i] = *(const uint32_t*)&As[row][kk];
                }
            }
            #pragma unroll
            for (int tn = 0; tn < 4; ++tn) {
                int col = wn + tn * 16 + lm;
                #pragma unroll
                for (int j = 0; j < 16; ++j)
                    fb[tn].s[j] = Bs[ks + ((j < 8) ? (lh * 8 + j) : (16 + lh * 8 + (j - 8)))][col];
            }
            #pragma unroll
            for (int tm = 0; tm < 2; ++tm)
                #pragma unroll
                for (int tn = 0; tn < 4; ++tn)
                    acc[tm][tn] = wmma_bf16(fa[tm], fb[tn], acc[tm][tn]);
        }
        __syncthreads();
    }

    #pragma unroll
    for (int tm = 0; tm < 2; ++tm)
        #pragma unroll
        for (int tn = 0; tn < 4; ++tn)
            #pragma unroll
            for (int r = 0; r < 8; ++r) {
                int row = bm + wm + tm * 16 + lh * 8 + r;
                int col = bn + wn + tn * 16 + lm;
                size_t idx = (size_t)row * N + col;
                float val = acc[tm][tn][r];
                if constexpr (MODE == 1) val += resid[idx];
                if constexpr (MODE == 2) Cb[idx] = f2bf(val);
                else                     Cf[idx] = val;
            }
}

// ---------------------------------------------------------------------------
// Flash attention: causal GQA. Block = 4 waves, each wave = 16 query rows.
// Q/K/V bf16; QK^T and P@V via v_wmma_f32_16x16x32_bf16; online softmax.
// K tile staged via async global->LDS; V staged transposed (scatter).
// ---------------------------------------------------------------------------
__global__ __launch_bounds__(128) void flash_attn_kernel(const unsigned short* __restrict__ Q,
                                                         const unsigned short* __restrict__ Kt,
                                                         const unsigned short* __restrict__ Vt,
                                                         unsigned short* __restrict__ O) {
    __shared__ unsigned short Ks[64][136];   // [kv][d]
    __shared__ unsigned short Vs[128][72];   // [d][kv] (transposed)
    __shared__ unsigned short Pst[4][16][32];

    const float scale = 0.08838834764831845f; // 1/sqrt(128)
    const float NEGM = -1e30f;

    int tid = threadIdx.x;
    int wv = tid >> 5, lane = tid & 31, lm = lane & 15, lh = lane >> 4;
    int qblk = blockIdx.x;   // 0..15 (64 rows each)
    int h    = blockIdx.y;   // 0..15
    int b    = blockIdx.z;   // 0..3
    int hkv  = h >> 2;
    int q0   = qblk * 64 + wv * 16;

    // Q tile -> 4 A-fragments (D split into 4 chunks of 32)
    Frag16 qf[4];
    {
        const unsigned short* qrow = Q + (((size_t)b * SS + q0 + lm) * HH + h) * DD;
        #pragma unroll
        for (int c = 0; c < 4; ++c)
            #pragma unroll
            for (int i = 0; i < 8; ++i) {
                int dd = c * 32 + ((i < 4) ? (lh * 8 + 2 * i) : (16 + lh * 8 + 2 * (i - 4)));
                qf[c].u[i] = *(const uint32_t*)(qrow + dd);
            }
    }

    v8f o[8];
    #pragma unroll
    for (int dt = 0; dt < 8; ++dt) { v8f z = {0.f,0.f,0.f,0.f,0.f,0.f,0.f,0.f}; o[dt] = z; }
    float mrow[8], lrow[8];
    #pragma unroll
    for (int r = 0; r < 8; ++r) { mrow[r] = NEGM; lrow[r] = 0.f; }

    int kvEnd = qblk * 64 + 64;
    for (int kc = 0; kc < kvEnd; kc += 64) {
        // ---- stage 64 K rows (async) and 64 V rows (transposed scatter) ----
        {
            int kvloc = tid & 63;
            int half  = tid >> 6;
            const unsigned short* ksrc = Kt + (((size_t)b * SS + kc + kvloc) * HKV + hkv) * DD;
            #pragma unroll
            for (int i2 = 0; i2 < 8; ++i2)
                async_b128(&Ks[kvloc][half * 64 + i2 * 8], ksrc + half * 64 + i2 * 8);
            const unsigned short* vsrc = Vt + (((size_t)b * SS + kc + kvloc) * HKV + hkv) * DD;
            #pragma unroll
            for (int i2 = 0; i2 < 64; ++i2)
                Vs[half * 64 + i2][kvloc] = vsrc[half * 64 + i2];
        }
        wait_async0();
        __syncthreads();

        #pragma unroll 1
        for (int kb = 0; kb < 2; ++kb) {
            int kvb = kb * 32;
            if (kc + kvb > q0 + 15) continue;   // wave-uniform causal skip

            v8f s0 = {0.f,0.f,0.f,0.f,0.f,0.f,0.f,0.f};
            v8f s1 = {0.f,0.f,0.f,0.f,0.f,0.f,0.f,0.f};
            #pragma unroll
            for (int c = 0; c < 4; ++c) {
                Frag16 kf0, kf1;
                #pragma unroll
                for (int i = 0; i < 8; ++i) {
                    int dd = c * 32 + ((i < 4) ? (lh * 8 + 2 * i) : (16 + lh * 8 + 2 * (i - 4)));
                    kf0.u[i] = *(const uint32_t*)&Ks[kvb + lm][dd];
                    kf1.u[i] = *(const uint32_t*)&Ks[kvb + 16 + lm][dd];
                }
                s0 = wmma_bf16(qf[c], kf0, s0);
                s1 = wmma_bf16(qf[c], kf1, s1);
            }

            #pragma unroll
            for (int r = 0; r < 8; ++r) {
                int qpos = q0 + lh * 8 + r;
                int col0 = kc + kvb + lm;
                int col1 = col0 + 16;
                bool v0 = (col0 <= qpos), v1 = (col1 <= qpos);
                float x0 = v0 ? s0[r] * scale : NEGM;
                float x1 = v1 ? s1[r] * scale : NEGM;
                float mx = fmaxf(x0, x1);
                mx = fmaxf(mx, __shfl_xor(mx, 1, 32));
                mx = fmaxf(mx, __shfl_xor(mx, 2, 32));
                mx = fmaxf(mx, __shfl_xor(mx, 4, 32));
                mx = fmaxf(mx, __shfl_xor(mx, 8, 32));
                float mnew  = fmaxf(mrow[r], mx);
                float alpha = __expf(mrow[r] - mnew);
                float p0 = v0 ? __expf(x0 - mnew) : 0.f;
                float p1 = v1 ? __expf(x1 - mnew) : 0.f;
                float ps = p0 + p1;
                ps += __shfl_xor(ps, 1, 32);
                ps += __shfl_xor(ps, 2, 32);
                ps += __shfl_xor(ps, 4, 32);
                ps += __shfl_xor(ps, 8, 32);
                lrow[r] = lrow[r] * alpha + ps;
                mrow[r] = mnew;
                #pragma unroll
                for (int dt = 0; dt < 8; ++dt) o[dt][r] *= alpha;
                Pst[wv][lh * 8 + r][lm]      = f2bf(p0);
                Pst[wv][lh * 8 + r][16 + lm] = f2bf(p1);
            }
            asm volatile("s_wait_dscnt 0" ::: "memory");

            Frag16 pf;
            #pragma unroll
            for (int i = 0; i < 8; ++i) {
                int kk = (i < 4) ? (lh * 8 + 2 * i) : (16 + lh * 8 + 2 * (i - 4));
                pf.u[i] = *(const uint32_t*)&Pst[wv][lm][kk];
            }
            #pragma unroll
            for (int dt = 0; dt < 8; ++dt) {
                Frag16 vf;
                #pragma unroll
                for (int i = 0; i < 8; ++i)
                    vf.u[i] = *(const uint32_t*)&Vs[dt * 16 + lm][kvb + ((i < 4) ? (lh * 8 + 2 * i) : (16 + lh * 8 + 2 * (i - 4)))];
                o[dt] = wmma_bf16(pf, vf, o[dt]);
            }
        }
        __syncthreads();
    }

    #pragma unroll
    for (int dt = 0; dt < 8; ++dt)
        #pragma unroll
        for (int r = 0; r < 8; ++r) {
            int qpos = q0 + lh * 8 + r;
            float val = o[dt][r] / lrow[r];
            O[(((size_t)b * SS + qpos) * HH + h) * DD + dt * 16 + lm] = f2bf(val);
        }
}

// ---------------------------------------------------------------------------
// Workspace layout (bytes, all 256-aligned)
// ---------------------------------------------------------------------------
static constexpr size_t OFF_H     = 0;
static constexpr size_t OFF_WQKV  = OFF_H     + (size_t)NTOK * HID * 2;
static constexpr size_t OFF_QKV   = OFF_WQKV  + (size_t)HID * QKVN * 2;
static constexpr size_t OFF_Q     = OFF_QKV   + (size_t)NTOK * QKVN * 4;
static constexpr size_t OFF_K     = OFF_Q     + (size_t)NTOK * HH * DD * 2;
static constexpr size_t OFF_V     = OFF_K     + (size_t)NTOK * HKV * DD * 2;
static constexpr size_t OFF_ATTN  = OFF_V     + (size_t)NTOK * HKV * DD * 2;
static constexpr size_t OFF_WO    = OFF_ATTN  + (size_t)NTOK * HID * 2;
static constexpr size_t OFF_H1    = OFF_WO    + (size_t)HID * HID * 2;
static constexpr size_t OFF_H2    = OFF_H1    + (size_t)NTOK * HID * 4;
static constexpr size_t OFF_WGU   = OFF_H2    + (size_t)NTOK * HID * 2;
static constexpr size_t OFF_GU    = OFF_WGU   + (size_t)HID * 2 * INTR * 2;
static constexpr size_t OFF_ACT   = OFF_GU    + (size_t)NTOK * 2 * INTR * 2;
static constexpr size_t OFF_WDOWN = OFF_ACT   + (size_t)NTOK * INTR * 2;
static constexpr size_t WS_NEEDED = OFF_WDOWN + (size_t)INTR * HID * 2;

extern "C" void kernel_launch(void* const* d_in, const int* in_sizes, int n_in,
                              void* d_out, int out_size, void* d_ws, size_t ws_size,
                              hipStream_t stream) {
    (void)in_sizes; (void)n_in; (void)out_size;
    if (ws_size < WS_NEEDED) return;

    const float* x       = (const float*)d_in[0];
    const float* ln1_w   = (const float*)d_in[1];
    const float* wqkv    = (const float*)d_in[2];
    const float* wo      = (const float*)d_in[3];
    const float* ln2_w   = (const float*)d_in[4];
    const float* wgu     = (const float*)d_in[5];
    const float* wdown   = (const float*)d_in[6];
    const int*   pos     = (const int*)d_in[7];
    float* out = (float*)d_out;

    char* ws = (char*)d_ws;
    unsigned short* h_b    = (unsigned short*)(ws + OFF_H);
    unsigned short* wqkv_b = (unsigned short*)(ws + OFF_WQKV);
    float*          qkv_f  = (float*)        (ws + OFF_QKV);
    unsigned short* q_b    = (unsigned short*)(ws + OFF_Q);
    unsigned short* k_b    = (unsigned short*)(ws + OFF_K);
    unsigned short* v_b    = (unsigned short*)(ws + OFF_V);
    unsigned short* attn_b = (unsigned short*)(ws + OFF_ATTN);
    unsigned short* wo_b   = (unsigned short*)(ws + OFF_WO);
    float*          h1_f   = (float*)        (ws + OFF_H1);
    unsigned short* h2_b   = (unsigned short*)(ws + OFF_H2);
    unsigned short* wgu_b  = (unsigned short*)(ws + OFF_WGU);
    unsigned short* gu_b   = (unsigned short*)(ws + OFF_GU);
    unsigned short* act_b  = (unsigned short*)(ws + OFF_ACT);
    unsigned short* wdn_b  = (unsigned short*)(ws + OFF_WDOWN);

    // Weight conversions fp32 -> bf16 (4-wide)
    convert_bf16_kernel<<<2048, 256, 0, stream>>>((const float4*)wqkv,  (uint32_t*)wqkv_b, (size_t)HID * QKVN / 4);
    convert_bf16_kernel<<<2048, 256, 0, stream>>>((const float4*)wo,    (uint32_t*)wo_b,   (size_t)HID * HID / 4);
    convert_bf16_kernel<<<2048, 256, 0, stream>>>((const float4*)wgu,   (uint32_t*)wgu_b,  (size_t)HID * 2 * INTR / 4);
    convert_bf16_kernel<<<2048, 256, 0, stream>>>((const float4*)wdown, (uint32_t*)wdn_b,  (size_t)INTR * HID / 4);

    // h = rmsnorm(x) -> bf16
    rmsnorm_kernel<<<NTOK, 256, 0, stream>>>(x, ln1_w, h_b);

    // qkv = h @ wqkv  (fp32 out)
    gemm_bf16_kernel<0><<<dim3(QKVN / 256, NTOK / 64), 256, 0, stream>>>(
        h_b, wqkv_b, NTOK, QKVN, HID, nullptr, qkv_f, nullptr);

    // RoPE + split to bf16 q/k/v
    rope_split_kernel<<<NTOK, 256, 0, stream>>>(qkv_f, pos, q_b, k_b, v_b);

    // attention -> bf16 [b, s, h, d]
    flash_attn_kernel<<<dim3(SS / 64, HH, BB), 128, 0, stream>>>(q_b, k_b, v_b, attn_b);

    // h1 = x + attn @ wo
    gemm_bf16_kernel<1><<<dim3(HID / 256, NTOK / 64), 256, 0, stream>>>(
        attn_b, wo_b, NTOK, HID, HID, x, h1_f, nullptr);

    // h2 = rmsnorm(h1) -> bf16
    rmsnorm_kernel<<<NTOK, 256, 0, stream>>>(h1_f, ln2_w, h2_b);

    // gu = h2 @ w_gate_up  (bf16 out)
    gemm_bf16_kernel<2><<<dim3((2 * INTR) / 256, NTOK / 64), 256, 0, stream>>>(
        h2_b, wgu_b, NTOK, 2 * INTR, HID, nullptr, nullptr, gu_b);

    // act = silu(gate) * up
    swiglu_kernel<<<8192, 256, 0, stream>>>(gu_b, act_b);

    // out = h1 + act @ w_down
    gemm_bf16_kernel<1><<<dim3(HID / 256, NTOK / 64), 256, 0, stream>>>(
        act_b, wdn_b, NTOK, HID, INTR, h1_f, out, nullptr);
}
